// MoleculeModel_48507360641147
// MI455X (gfx1250) — matrix operand order repaired and checked by reading, not compiled, not run on hardware
//
#include <hip/hip_runtime.h>

typedef __attribute__((ext_vector_type(16))) __bf16 v16bf;
typedef __attribute__((ext_vector_type(8)))  float  v8f;

#define FDIM 512
#define KT   16   // 512 / 32 (WMMA K)
#define GT   32   // 512 / 16 (WMMA N)

// ---------------------------------------------------------------------------
// Pre-pass: convert f32 Q (upper-triangular, row-major [512,512]) into bf16
// WMMA B-fragment layout in workspace: frag[kt][gt][lane][16 elems].
// B (32x16 bf16) in contiguous K order (A matches): lane n (0..15) holds column n, K=0..15;
// lane n+16 holds column n, K=16..31; 16 bf16 (=32B) contiguous per lane.
// One thread per (frag, lane) -> 16*32*32 = 16384 threads, 32B stores.
// ---------------------------------------------------------------------------
__global__ __launch_bounds__(256) void pack_q_bf16(
    const float* __restrict__ Q, __bf16* __restrict__ qp) {
  int t = blockIdx.x * blockDim.x + threadIdx.x;
  if (t >= KT * GT * 32) return;
  int lane = t & 31;
  int gt   = (t >> 5) & 31;
  int kt   = t >> 10;
  int n  = gt * 16 + (lane & 15);
  int k0 = kt * 32 + ((lane >> 4) ? 16 : 0);
  v16bf v;
#pragma unroll
  for (int e = 0; e < 16; ++e)
    v[e] = (__bf16)Q[(size_t)(k0 + e) * FDIM + n];
  *(v16bf*)(qp + (size_t)t * 16) = v;
}

// ---------------------------------------------------------------------------
// Main kernel: one wave32 per 16-row slab of x.
// out[m] = sum_g (x@Q)[m,g] * x[m,g], with triangular k-tile skipping.
// ---------------------------------------------------------------------------
__global__ __launch_bounds__(256) void bilinear_diag_wmma(
    const float* __restrict__ x,
    const __bf16* __restrict__ qp,
    float* __restrict__ out) {
  const int lane = threadIdx.x & 31;
  const int wave = threadIdx.x >> 5;
  const int m0   = (blockIdx.x * (blockDim.x >> 5) + wave) * 16;
  const int hi   = lane >> 4;   // half-wave select
  const int ln   = lane & 15;

  // A-fragment source row for this lane (row M = ln)
  const float* xrow = x + (size_t)(m0 + ln) * FDIM;

  v8f acc = {};  // acc[r] ~ partial out for row m0 + r + 8*hi

  for (int gt = 0; gt < GT; ++gt) {
    v8f c = {};
    const int ktmax = gt >> 1;  // Q upper-triangular: tiles with kt > gt/2 are zero
    for (int kt = 0; kt <= ktmax; ++kt) {
      // --- A fragment (16x32 bf16): lane<16 -> K {k..k+15}, lane>=16 -> K {k+16..k+31} of 32-chunk
      const int k0 = kt * 32 + hi * 16;
      const float4* p = (const float4*)(xrow + k0);
      float4 a0 = p[0], a1 = p[1];   // K = k0 .. k0+7
      float4 a2 = p[2], a3 = p[3];   // K = k0+8 .. k0+15
      v16bf a;
      a[0]  = (__bf16)a0.x; a[1]  = (__bf16)a0.y;
      a[2]  = (__bf16)a0.z; a[3]  = (__bf16)a0.w;
      a[4]  = (__bf16)a1.x; a[5]  = (__bf16)a1.y;
      a[6]  = (__bf16)a1.z; a[7]  = (__bf16)a1.w;
      a[8]  = (__bf16)a2.x; a[9]  = (__bf16)a2.y;
      a[10] = (__bf16)a2.z; a[11] = (__bf16)a2.w;
      a[12] = (__bf16)a3.x; a[13] = (__bf16)a3.y;
      a[14] = (__bf16)a3.z; a[15] = (__bf16)a3.w;

      // --- B fragment: prepacked, fully coalesced 32B per lane (L2-hot)
      const v16bf b = *(const v16bf*)(qp + ((size_t)(kt * GT + gt) * 32 + lane) * 16);

      c = __builtin_amdgcn_wmma_f32_16x16x32_bf16(
              false, a, false, b, (short)0, c, false, false);
    }
    // Fold y[m,g] * x[m,g] into persistent accumulator.
    // C/D layout: VGPR r, lane l -> (M = r + 8*(l>>4), N = gt*16 + (l&15)).
    const int g = gt * 16 + ln;
#pragma unroll
    for (int r = 0; r < 8; ++r) {
      float xv = x[(size_t)(m0 + r + 8 * hi) * FDIM + g];
      acc[r] += c[r] * xv;
    }
  }

  // Reduce over the 16 lanes of each half-wave (columns), then store.
#pragma unroll
  for (int r = 0; r < 8; ++r) {
    float v = acc[r];
    v += __shfl_xor(v, 1, 32);
    v += __shfl_xor(v, 2, 32);
    v += __shfl_xor(v, 4, 32);
    v += __shfl_xor(v, 8, 32);
    if (ln == 0) out[m0 + r + 8 * hi] = v;
  }
}

// ---------------------------------------------------------------------------
extern "C" void kernel_launch(void* const* d_in, const int* in_sizes, int n_in,
                              void* d_out, int out_size, void* d_ws, size_t ws_size,
                              hipStream_t stream) {
  const float* x = (const float*)d_in[0];   // [B, 512] f32
  const float* Q = (const float*)d_in[1];   // [512, 512] f32, upper-triangular
  float* out = (float*)d_out;               // [B, 1] f32
  __bf16* qp = (__bf16*)d_ws;               // 512 KB packed Q (bf16)

  // 1) Pack Q into bf16 B-fragment layout (16384 threads).
  pack_q_bf16<<<(KT * GT * 32 + 255) / 256, 256, 0, stream>>>(Q, qp);

  // 2) Bilinear diagonal: 8 waves/block -> 128 rows/block.
  const int B = in_sizes[0] / FDIM;         // 131072
  const int blocks = B / 128;               // 1024
  bilinear_diag_wmma<<<blocks, 256, 0, stream>>>(x, qp, out);
}
